// Model_45329084842151
// MI455X (gfx1250) — matrix operand order repaired and checked by reading, not compiled, not run on hardware
//
#include <hip/hip_runtime.h>
#include <hip/hip_bf16.h>

// ---------------- problem constants (from reference) ----------------
#define N_      8192
#define S_      24
#define C_      4
#define E_      65536
#define WIN_    20
#define NW_     4            // S - WIN
#define NF_     2
#define K_      5
#define Y_      128          // C*32 (last GCN layer width)
#define FLAT_   2560         // WIN*Y
#define INSZ_   2560         // NF*FLAT/2
#define PROJ_   1280
#define HID_    128
#define LBL_    12

typedef _Float16 v16h __attribute__((ext_vector_type(16)));
typedef _Float16 v8h  __attribute__((ext_vector_type(8)));
typedef float    v8f  __attribute__((ext_vector_type(8)));

// ---------------- utility kernels ----------------
__global__ void zero_f32(float* p, long long n) {
    long long i = (long long)blockIdx.x * blockDim.x + threadIdx.x;
    if (i < n) p[i] = 0.0f;
}

__global__ void f32_to_f16(const float* __restrict__ src, _Float16* __restrict__ dst, long long n) {
    long long i = (long long)blockIdx.x * blockDim.x + threadIdx.x;
    if (i < n) dst[i] = (_Float16)src[i];
}

// ---------------- graph normalization ----------------
__global__ void deg_kernel(const int* __restrict__ ei, const float* __restrict__ ea,
                           float* __restrict__ deg) {
    int e = blockIdx.x * blockDim.x + threadIdx.x;
    if (e >= E_) return;
    atomicAdd(&deg[ei[E_ + e]], ea[e]);
}

__global__ void dinv_kernel(const float* __restrict__ deg, float* __restrict__ dinv) {
    int n = blockIdx.x * blockDim.x + threadIdx.x;
    if (n >= N_) return;
    float d = deg[n] + 1.0f;   // + self loop weight
    dinv[n] = (d > 0.0f) ? rsqrtf(d) : 0.0f;
}

__global__ void norm_kernel(const int* __restrict__ ei, const float* __restrict__ ea,
                            const float* __restrict__ dinv, float* __restrict__ nrm) {
    int e = blockIdx.x * blockDim.x + threadIdx.x;
    if (e >= E_) return;
    nrm[e] = dinv[ei[e]] * ea[e] * dinv[ei[E_ + e]];
}

// ---------------- GCN layer pieces ----------------
// h[row, o] = sum_c x[row, c] * W[o, c]   (rows = N*S; small ci -> VALU is fine)
__global__ void gcn_transform(const float* __restrict__ x, const float* __restrict__ W,
                              float* __restrict__ h, int rows, int ci, int co) {
    long long i = (long long)blockIdx.x * blockDim.x + threadIdx.x;
    if (i >= (long long)rows * co) return;
    int row = (int)(i / co), o = (int)(i % co);
    const float* xr = x + (size_t)row * ci;
    const float* wr = W + (size_t)o * ci;
    float s = 0.0f;
    for (int c = 0; c < ci; ++c) s += xr[c] * wr[c];
    h[i] = s;
}

// thread per (edge, s, o/4) : agg[dst,s,o..o+3] += norm[e]*h[src,s,o..o+3]
__global__ void gcn_scatter(const float* __restrict__ h, const int* __restrict__ ei,
                            const float* __restrict__ nrm, float* __restrict__ agg, int co) {
    int q = co >> 2;
    long long i = (long long)blockIdx.x * blockDim.x + threadIdx.x;
    if (i >= (long long)E_ * S_ * q) return;
    int e   = (int)(i / ((long long)S_ * q));
    int rem = (int)(i % ((long long)S_ * q));
    int s   = rem / q;
    int o4  = (rem % q) << 2;
    int src = ei[e], dst = ei[E_ + e];
    float nm = nrm[e];
    const float4 hv = *(const float4*)(h + ((size_t)src * S_ + s) * co + o4);
    float* ap = agg + ((size_t)dst * S_ + s) * co + o4;
    atomicAdd(ap + 0, nm * hv.x);
    atomicAdd(ap + 1, nm * hv.y);
    atomicAdd(ap + 2, nm * hv.z);
    atomicAdd(ap + 3, nm * hv.w);
}

// x_out = tanh(agg + dinv[n]^2 * h + b)   (self-loop folded in here)
__global__ void gcn_finish(const float* __restrict__ agg, const float* __restrict__ h,
                           const float* __restrict__ dinv, const float* __restrict__ b,
                           float* __restrict__ xout, int co) {
    long long i = (long long)blockIdx.x * blockDim.x + threadIdx.x;
    if (i >= (long long)N_ * S_ * co) return;
    int n = (int)(i / ((long long)S_ * co));
    int o = (int)(i % co);
    float d = dinv[n];
    xout[i] = tanhf(agg[i] + d * d * h[i] + b[o]);
}

// ---------------- window + conv1d(SAME,K=5) + pairwise maxpool + f16 quantize ----------------
__global__ void conv_pool_kernel(const float* __restrict__ xf,   // N x S x Y
                                 const float* __restrict__ cw,   // NF x 1 x K
                                 const float* __restrict__ cb,   // NF
                                 _Float16* __restrict__ Af16) {  // (N*NW) x INSZ
    long long i = (long long)blockIdx.x * blockDim.x + threadIdx.x;
    if (i >= (long long)N_ * NW_ * INSZ_) return;
    int p  = (int)(i % INSZ_);
    long long nw = i / INSZ_;
    int w = (int)(nw % NW_);
    int n = (int)(nw / NW_);
    int flatbase = 2 * p;
    int f   = flatbase / FLAT_;
    int pos = flatbase % FLAT_;
    const float* xrow = xf + (size_t)n * S_ * Y_;
    float best = -3.4e38f;
    #pragma unroll
    for (int pair = 0; pair < 2; ++pair) {
        int pp = pos + pair;
        float acc = cb[f];
        #pragma unroll
        for (int k = 0; k < K_; ++k) {
            int xi = pp + k - 2;                 // SAME padding (pad 2 each side)
            if (xi >= 0 && xi < FLAT_) {
                int t  = w + (xi >> 7);          // Y_ == 128
                int ch = xi & 127;
                acc += xrow[t * Y_ + ch] * cw[f * K_ + k];
            }
        }
        best = fmaxf(best, acc);
    }
    Af16[i] = (_Float16)best;
}

// ---------------- WMMA f16 GEMM: LDS double-buffered, 2x2 register-blocked ----------------
// C(MxN) = A(MxK) * B(NxK)^T [+ bias].  B is the weight matrix in its natural
// (N,K) row-major layout (WMMA B fragments want contiguous K per lane).
// Block: 256 threads = 8 waves -> 128(M) x 64(N) tile; wave: 32x32 = 4 accumulators.
// Pipeline per 64-K round: issue next-tile global b128 loads -> 8 WMMAs from current
// LDS buffer -> ds_store next tile into other buffer -> one barrier.
// Requires M%128==0, N%64==0, K%64==0 (true for all three call sites).
#define LDT_ 72           // LDS row stride in f16 (144B: 16B-aligned, conflict-free)
#define ASZ_ (128 * LDT_) // one A buffer, f16 elements
#define BSZ_ (64 * LDT_)  // one B buffer, f16 elements

__global__ __launch_bounds__(256) void wmma_gemm_f16(
        const _Float16* __restrict__ A,   // M x K
        const _Float16* __restrict__ Bw,  // N x K
        const float* __restrict__ bias,   // length N or nullptr
        float* __restrict__ Cout,         // M x N
        int M, int Nn, int K) {
    __shared__ _Float16 As[2 * ASZ_];
    __shared__ _Float16 Bs[2 * BSZ_];

    const int tid  = threadIdx.x;
    const int lane = tid & 31;
    const int wave = tid >> 5;
    const int wm   = (wave & 3) * 32;   // wave M offset in block tile
    const int wn   = (wave >> 2) * 32;  // wave N offset in block tile
    const int g    = lane >> 4;         // lane group
    const int r    = lane & 15;
    const int mB   = blockIdx.x * 128;
    const int nB   = blockIdx.y * 64;

    // cooperative staging assignments (KB = 64 per round)
    const int arow = tid >> 1;          // 0..127, 2 threads/row, 32 f16 each
    const int akq  = (tid & 1) * 32;
    const int brow = tid >> 2;          // 0..63, 4 threads/row, 16 f16 each
    const int bkq  = (tid & 3) * 16;

    const _Float16* Ag = A  + (size_t)(mB + arow) * K + akq;
    const _Float16* Bg = Bw + (size_t)(nB + brow) * K + bkq;
    const int aswo = arow * LDT_ + akq;          // LDS store offsets (within a buffer)
    const int bswo = brow * LDT_ + bkq;
    const int apo0 = (wm + r) * LDT_ + 16 * g;   // LDS frag offsets (within a buffer)
    const int bpo0 = (wn + r) * LDT_ + 16 * g;

    v8f acc00 = {}, acc01 = {}, acc10 = {}, acc11 = {};
    v8h sa0, sa1, sa2, sa3, sb0, sb1;            // staging registers

    // prologue: stage tile 0 into buffer 0
    sa0 = *(const v8h*)(Ag);      sa1 = *(const v8h*)(Ag + 8);
    sa2 = *(const v8h*)(Ag + 16); sa3 = *(const v8h*)(Ag + 24);
    sb0 = *(const v8h*)(Bg);      sb1 = *(const v8h*)(Bg + 8);
    *(v8h*)(As + aswo)      = sa0; *(v8h*)(As + aswo + 8)  = sa1;
    *(v8h*)(As + aswo + 16) = sa2; *(v8h*)(As + aswo + 24) = sa3;
    *(v8h*)(Bs + bswo)      = sb0; *(v8h*)(Bs + bswo + 8)  = sb1;
    __syncthreads();

    const int KT = K >> 6;   // number of 64-wide K rounds
    for (int kt = 0; kt < KT; ++kt) {
        const int cur = kt & 1;
        const bool more = (kt + 1 < KT);

        // (1) issue next tile's global loads (latency hides under the WMMAs below)
        if (more) {
            const int kn = (kt + 1) * 64;
            sa0 = *(const v8h*)(Ag + kn);      sa1 = *(const v8h*)(Ag + kn + 8);
            sa2 = *(const v8h*)(Ag + kn + 16); sa3 = *(const v8h*)(Ag + kn + 24);
            sb0 = *(const v8h*)(Bg + kn);      sb1 = *(const v8h*)(Bg + kn + 8);
            __builtin_prefetch(Ag + kn + 64, 0, 0);    // global_prefetch_b8
            __builtin_prefetch(Bg + kn + 64, 0, 0);
        }

        // (2) compute from current LDS buffer
        const _Float16* Ab = As + cur * ASZ_;
        const _Float16* Bb = Bs + cur * BSZ_;
        const _Float16* ap0 = Ab + apo0;
        const _Float16* ap1 = ap0 + 16 * LDT_;
        const _Float16* bp0 = Bb + bpo0;
        const _Float16* bp1 = bp0 + 16 * LDT_;
        #pragma unroll
        for (int ks = 0; ks < 64; ks += 32) {
            // A fragments: elems 0..15 = K(ks+16g..+15), contiguous
            v8h a0lo = *(const v8h*)(ap0 + ks);
            v8h a0hi = *(const v8h*)(ap0 + ks + 8);
            v8h a1lo = *(const v8h*)(ap1 + ks);
            v8h a1hi = *(const v8h*)(ap1 + ks + 8);
            // B fragments: elems 0..15 = K(ks+16g..+15), contiguous
            v8h b0lo = *(const v8h*)(bp0 + ks);
            v8h b0hi = *(const v8h*)(bp0 + ks + 8);
            v8h b1lo = *(const v8h*)(bp1 + ks);
            v8h b1hi = *(const v8h*)(bp1 + ks + 8);
            v16h a0 = __builtin_shufflevector(a0lo, a0hi, 0,1,2,3,4,5,6,7,8,9,10,11,12,13,14,15);
            v16h a1 = __builtin_shufflevector(a1lo, a1hi, 0,1,2,3,4,5,6,7,8,9,10,11,12,13,14,15);
            v16h b0 = __builtin_shufflevector(b0lo, b0hi, 0,1,2,3,4,5,6,7,8,9,10,11,12,13,14,15);
            v16h b1 = __builtin_shufflevector(b1lo, b1hi, 0,1,2,3,4,5,6,7,8,9,10,11,12,13,14,15);
            acc00 = __builtin_amdgcn_wmma_f32_16x16x32_f16(false, a0, false, b0, (short)0, acc00, false, false);
            acc01 = __builtin_amdgcn_wmma_f32_16x16x32_f16(false, a0, false, b1, (short)0, acc01, false, false);
            acc10 = __builtin_amdgcn_wmma_f32_16x16x32_f16(false, a1, false, b0, (short)0, acc10, false, false);
            acc11 = __builtin_amdgcn_wmma_f32_16x16x32_f16(false, a1, false, b1, (short)0, acc11, false, false);
        }

        // (3) store next tile into the other buffer; (4) one barrier per round
        if (more) {
            _Float16* Aw = As + (cur ^ 1) * ASZ_;
            _Float16* Bww = Bs + (cur ^ 1) * BSZ_;
            *(v8h*)(Aw + aswo)      = sa0; *(v8h*)(Aw + aswo + 8)  = sa1;
            *(v8h*)(Aw + aswo + 16) = sa2; *(v8h*)(Aw + aswo + 24) = sa3;
            *(v8h*)(Bww + bswo)     = sb0; *(v8h*)(Bww + bswo + 8) = sb1;
        }
        __syncthreads();
    }

    const int m0 = mB + wm;
    const int n0 = nB + wn;
    float bv0 = bias ? bias[n0 + r]      : 0.0f;
    float bv1 = bias ? bias[n0 + 16 + r] : 0.0f;
    // D layout: VGPR vr holds row (tile_m + vr + 8*g), col (tile_n + r)
    #pragma unroll
    for (int vr = 0; vr < 8; ++vr) {
        int ra = m0 + vr + 8 * g;
        int rb = ra + 16;
        Cout[(size_t)ra * Nn + n0 + r]      = acc00[vr] + bv0;
        Cout[(size_t)ra * Nn + n0 + 16 + r] = acc01[vr] + bv1;
        Cout[(size_t)rb * Nn + n0 + r]      = acc10[vr] + bv0;
        Cout[(size_t)rb * Nn + n0 + 16 + r] = acc11[vr] + bv1;
    }
}

// ---------------- LSTM gates ----------------
__device__ __forceinline__ float sigf(float x) { return 1.0f / (1.0f + expf(-x)); }

__global__ void lstm_gate(const float* __restrict__ gih,   // (N*NW) x 4H
                          const float* __restrict__ ghh,   // N x 4H
                          const float* __restrict__ b_ih, const float* __restrict__ b_hh,
                          float* __restrict__ cst, float* __restrict__ hst,
                          _Float16* __restrict__ hf16, int t) {
    int i = blockIdx.x * blockDim.x + threadIdx.x;
    if (i >= N_ * HID_) return;
    int n = i / HID_, j = i % HID_;
    const float* gi = gih + ((size_t)n * NW_ + t) * (4 * HID_);
    const float* gh = ghh + (size_t)n * (4 * HID_);
    float ii = gi[j]             + gh[j]             + b_ih[j]             + b_hh[j];
    float ff = gi[HID_ + j]      + gh[HID_ + j]      + b_ih[HID_ + j]      + b_hh[HID_ + j];
    float gg = gi[2 * HID_ + j]  + gh[2 * HID_ + j]  + b_ih[2 * HID_ + j]  + b_hh[2 * HID_ + j];
    float oo = gi[3 * HID_ + j]  + gh[3 * HID_ + j]  + b_ih[3 * HID_ + j]  + b_hh[3 * HID_ + j];
    float cn = sigf(ff) * cst[i] + sigf(ii) * tanhf(gg);
    float hn = sigf(oo) * tanhf(cn);
    cst[i] = cn;
    hst[i] = hn;
    hf16[i] = (_Float16)hn;
}

// ---------------- output head ----------------
__global__ void out_kernel(const float* __restrict__ h, const float* __restrict__ ow,
                           const float* __restrict__ ob, float* __restrict__ out) {
    int i = blockIdx.x * blockDim.x + threadIdx.x;
    if (i >= N_ * LBL_) return;
    int n = i / LBL_, l = i % LBL_;
    const float* hr = h + (size_t)n * HID_;
    const float* wr = ow + (size_t)l * HID_;
    float s = ob[l];
    #pragma unroll 8
    for (int j = 0; j < HID_; ++j) s += hr[j] * wr[j];
    out[i] = s;
}

// ---------------- host launcher ----------------
static inline unsigned gsz(long long total, int bs) { return (unsigned)((total + bs - 1) / bs); }

extern "C" void kernel_launch(void* const* d_in, const int* in_sizes, int n_in,
                              void* d_out, int out_size, void* d_ws, size_t ws_size,
                              hipStream_t stream) {
    (void)in_sizes; (void)n_in; (void)out_size; (void)ws_size;

    const float* inputs = (const float*)d_in[0];
    const int*   ei     = (const int*)d_in[1];
    const float* ea     = (const float*)d_in[2];
    const float* Wl[5]  = {(const float*)d_in[3], (const float*)d_in[5], (const float*)d_in[7],
                           (const float*)d_in[9], (const float*)d_in[11]};
    const float* Bl[5]  = {(const float*)d_in[4], (const float*)d_in[6], (const float*)d_in[8],
                           (const float*)d_in[10], (const float*)d_in[12]};
    const float* conv_w = (const float*)d_in[13];
    const float* conv_b = (const float*)d_in[14];
    const float* proj_w = (const float*)d_in[15];
    const float* proj_b = (const float*)d_in[16];
    const float* w_ih   = (const float*)d_in[17];
    const float* w_hh   = (const float*)d_in[18];
    const float* b_ih   = (const float*)d_in[19];
    const float* b_hh   = (const float*)d_in[20];
    const float* out_w  = (const float*)d_in[21];
    const float* out_b  = (const float*)d_in[22];
    float* out = (float*)d_out;

    // ---- workspace layout (lifetime-overlapped regions) ----
    char* ws = (char*)d_ws;
    size_t off = 0;
    auto carve = [&](size_t bytes) { size_t o = off; off += (bytes + 255) & ~(size_t)255; return o; };

    float*    deg    = (float*)(ws + carve((size_t)N_ * 4));
    float*    dinv   = (float*)(ws + carve((size_t)N_ * 4));
    float*    nrm    = (float*)(ws + carve((size_t)E_ * 4));
    _Float16* projwB = (_Float16*)(ws + carve((size_t)PROJ_ * INSZ_ * 2));      // (N=1280, K=2560)
    _Float16* wihB   = (_Float16*)(ws + carve((size_t)4 * HID_ * PROJ_ * 2));   // (N=512,  K=1280)
    _Float16* whhB   = (_Float16*)(ws + carve((size_t)4 * HID_ * HID_ * 2));    // (N=512,  K=128)

    const size_t SZ_X = (size_t)N_ * S_ * 128 * 4;   // ~100 MB, max GCN activation
    size_t r1 = carve(SZ_X);                         // GCN x  | later seq_f16
    size_t r2 = carve(SZ_X);                         // GCN h  | later Af16 (spans r2+r3) | later gates_ih
    size_t r3 = carve(SZ_X);                         // GCN agg
    size_t r4 = carve((size_t)N_ * NW_ * PROJ_ * 4); // seq f32 (~168 MB) | later LSTM state

    float*    xbuf   = (float*)(ws + r1);
    float*    hbuf   = (float*)(ws + r2);
    float*    agg    = (float*)(ws + r3);
    _Float16* Af16   = (_Float16*)(ws + r2);                     // 168 MB, fits in r2+r3
    float*    seq    = (float*)(ws + r4);
    _Float16* seqf16 = (_Float16*)(ws + r1);                     // 84 MB
    float*    gih    = (float*)(ws + r2);                        // 67 MB
    float*    ghh    = (float*)(ws + r4);                        // 16 MB (seq dead by then)
    float*    cst    = (float*)(ws + r4 + ((size_t)N_ * 4 * HID_ * 4));
    float*    hst    = (float*)(ws + r4 + ((size_t)N_ * 4 * HID_ * 4) + (size_t)N_ * HID_ * 4);
    _Float16* hf16   = (_Float16*)(ws + r4 + ((size_t)N_ * 4 * HID_ * 4) + 2 * (size_t)N_ * HID_ * 4);

    const int BS = 256;

    // ---- graph normalization ----
    zero_f32<<<gsz(N_, BS), BS, 0, stream>>>(deg, N_);
    deg_kernel<<<gsz(E_, BS), BS, 0, stream>>>(ei, ea, deg);
    dinv_kernel<<<gsz(N_, BS), BS, 0, stream>>>(deg, dinv);
    norm_kernel<<<gsz(E_, BS), BS, 0, stream>>>(ei, ea, dinv, nrm);

    // ---- 5 GCN layers (memory/atomic-bound -> VALU + f32 atomics) ----
    const int cis[5] = {4, 8, 16, 32, 64};
    const int cos_[5] = {8, 16, 32, 64, 128};
    const float* xin = inputs;
    for (int l = 0; l < 5; ++l) {
        int ci = cis[l], co = cos_[l];
        long long rows = (long long)N_ * S_;
        gcn_transform<<<gsz(rows * co, BS), BS, 0, stream>>>(xin, Wl[l], hbuf, (int)rows, ci, co);
        zero_f32<<<gsz(rows * co, BS), BS, 0, stream>>>(agg, rows * co);
        gcn_scatter<<<gsz((long long)E_ * S_ * (co / 4), BS), BS, 0, stream>>>(hbuf, ei, nrm, agg, co);
        gcn_finish<<<gsz(rows * co, BS), BS, 0, stream>>>(agg, hbuf, dinv, Bl[l], xbuf, co);
        xin = xbuf;
    }

    // ---- window + conv + pool -> f16 A matrix (32768 x 2560) ----
    conv_pool_kernel<<<gsz((long long)N_ * NW_ * INSZ_, BS), BS, 0, stream>>>(xbuf, conv_w, conv_b, Af16);

    // ---- weight f16 conversion (natural (N,K) layout; no transpose needed) ----
    f32_to_f16<<<gsz((long long)PROJ_ * INSZ_, BS), BS, 0, stream>>>(proj_w, projwB, (long long)PROJ_ * INSZ_);
    f32_to_f16<<<gsz((long long)4 * HID_ * PROJ_, BS), BS, 0, stream>>>(w_ih, wihB, (long long)4 * HID_ * PROJ_);
    f32_to_f16<<<gsz((long long)4 * HID_ * HID_, BS), BS, 0, stream>>>(w_hh, whhB, (long long)4 * HID_ * HID_);

    // ---- projection GEMM (215 GFLOP, compute-bound -> WMMA) ----
    {
        dim3 grid((N_ * NW_) / 128, PROJ_ / 64);
        wmma_gemm_f16<<<grid, 256, 0, stream>>>(Af16, projwB, proj_b, seq, N_ * NW_, PROJ_, INSZ_);
    }

    // ---- batched LSTM input GEMM for all 4 steps (WMMA) ----
    f32_to_f16<<<gsz((long long)N_ * NW_ * PROJ_, BS), BS, 0, stream>>>(seq, seqf16, (long long)N_ * NW_ * PROJ_);
    {
        dim3 grid((N_ * NW_) / 128, (4 * HID_) / 64);
        wmma_gemm_f16<<<grid, 256, 0, stream>>>(seqf16, wihB, nullptr, gih, N_ * NW_, 4 * HID_, PROJ_);
    }

    // ---- LSTM recurrence (hh GEMM via WMMA each step) ----
    zero_f32<<<gsz((long long)N_ * 4 * HID_, BS), BS, 0, stream>>>(ghh, (long long)N_ * 4 * HID_);
    zero_f32<<<gsz((long long)N_ * HID_, BS), BS, 0, stream>>>(cst, (long long)N_ * HID_);
    for (int t = 0; t < NW_; ++t) {
        if (t > 0) {
            dim3 grid(N_ / 128, (4 * HID_) / 64);
            wmma_gemm_f16<<<grid, 256, 0, stream>>>(hf16, whhB, nullptr, ghh, N_, 4 * HID_, HID_);
        }
        lstm_gate<<<gsz((long long)N_ * HID_, BS), BS, 0, stream>>>(gih, ghh, b_ih, b_hh, cst, hst, hf16, t);
    }

    // ---- output head ----
    out_kernel<<<gsz((long long)N_ * LBL_, BS), BS, 0, stream>>>(hst, out_w, out_b, out);
}
